// QTARRouter_21474836480752
// MI455X (gfx1250) — matrix operand order repaired and checked by reading, not compiled, not run on hardware
//
#include <hip/hip_runtime.h>
#include <hip/hip_bf16.h>

// ---------------------------------------------------------------------------
// Problem constants (match reference)
// ---------------------------------------------------------------------------
constexpr int N_NODES  = 50000;
constexpr int N_EDGES  = 800000;
constexpr int HID      = 64;
constexpr int KEEP     = 400000;          // min(max(64, 0.5*800000), 800000)
constexpr int NTILES   = N_EDGES / 16;    // 50000 exactly (no tail)
constexpr int NPAD     = 1 << 20;         // bitonic sort size (>= N_EDGES)
constexpr int W1_STR   = 264;             // bf16 elems per row in LDS (16B pad)
constexpr int SORT_TILE = 4096;           // elems per block in LDS sort phases

// d_out layout (elements, 4B each):
//   [0, 1.2M)     kept_edges (int32 400000x3) | [1.2M,1.6M) kept_batch (int32)
//   [1.6M, 2.0M)  kept_weights (f32)          | [2.0M,2.05M) new_evidence (f32)
constexpr int OFF_BATCH = KEEP * 3;
constexpr int OFF_W     = OFF_BATCH + KEEP;
constexpr int OFF_EV    = OFF_W + KEEP;

typedef __bf16 v16bf __attribute__((ext_vector_type(16)));
typedef float  v8f   __attribute__((ext_vector_type(8)));
typedef unsigned long long u64;

// fp32 pair -> packed bf16x2 (1 VALU op): hardware cvt_pk if available,
// else single v_perm_b32 byte-pack (truncation).
__device__ __forceinline__ unsigned pk2(float lo, float hi) {
#if __has_builtin(__builtin_amdgcn_cvt_pk_bf16_f32)
    typedef __bf16 v2bf __attribute__((ext_vector_type(2)));
    union { v2bf v; unsigned u; } c;
    c.v = __builtin_amdgcn_cvt_pk_bf16_f32(lo, hi);
    return c.u;
#else
    return __builtin_amdgcn_perm(__float_as_uint(hi), __float_as_uint(lo),
                                 0x07060302u);
#endif
}

// ---------------------------------------------------------------------------
// Kernel 1: per-edge MLP gate via bf16 WMMA. One wave = one 16-edge tile.
//   feat(16x256) @ W1(256x64) -> relu -> dot W2 -> sigmoid -> score
// ---------------------------------------------------------------------------
__global__ __launch_bounds__(256) void score_wmma_kernel(
    const float* __restrict__ hidden, const int* __restrict__ edges,
    const int* __restrict__ q_rel,    const int* __restrict__ ebi,
    const float* __restrict__ evidence, const float* __restrict__ rel_embed,
    const float* __restrict__ W1, const float* __restrict__ b1,
    const float* __restrict__ W2, const float* __restrict__ b2,
    u64* __restrict__ keys)
{
    // W1 staged transposed (n-major) bf16, padded stride to spread LDS banks.
    __shared__ __align__(16) unsigned short sW1[64 * W1_STR];
    for (int idx = threadIdx.x; idx < 64 * 128; idx += blockDim.x) {
        int n = idx >> 7, kp = (idx & 127) * 2;
        unsigned p = pk2(W1[kp * 64 + n], W1[(kp + 1) * 64 + n]);
        *(unsigned*)(&sW1[n * W1_STR + kp]) = p;
    }
    __syncthreads();

    const int lane  = threadIdx.x & 31;
    const int lsub  = lane & 15;      // edge-in-tile this lane gathers
    const int hi    = lane >> 4;      // half-wave select
    const int wave  = blockIdx.x * (blockDim.x >> 5) + (threadIdx.x >> 5);
    const int nwave = gridDim.x * (blockDim.x >> 5);
    const float b2v = b2[0];
    const int kb  = hi * 8;           // A layout: lanes16-31 own K+8..15,+24..31
    const int kb2 = hi * 16;          // B layout: same as A (K+8..15,+24..31 for lanes16-31)

    for (int tile = wave; tile < NTILES; tile += nwave) {
        const int eb = tile * 16;
        const int e  = eb + lsub;
        __builtin_prefetch(edges + (size_t)3 * (e + 16 * nwave), 0, 0);

        const int sub = edges[3 * e + 0];
        const int rel = edges[3 * e + 1];
        const int obj = edges[3 * e + 2];
        const int q   = q_rel[ebi[e]];
        const float* bases[4] = { hidden + (size_t)sub * HID,
                                  hidden + (size_t)obj * HID,
                                  rel_embed + (size_t)rel * HID,
                                  rel_embed + (size_t)q   * HID };

        v8f acc[4] = {{}, {}, {}, {}};
        const unsigned short* brow = sW1 + lsub * W1_STR + kb;

        #pragma unroll
        for (int kc = 0; kc < 8; ++kc) {
            // A fragment: K = kc*32 + {kb..kb+7, kb+16..kb+23} of the feat row
            const float* p = bases[kc >> 1] + ((kc & 1) * 32 + kb);
            float4 q0 = *(const float4*)(p);
            float4 q1 = *(const float4*)(p + 4);
            float4 q2 = *(const float4*)(p + 16);
            float4 q3 = *(const float4*)(p + 20);
            union { v16bf v; unsigned u[8]; } A;
            A.u[0] = pk2(q0.x, q0.y); A.u[1] = pk2(q0.z, q0.w);
            A.u[2] = pk2(q1.x, q1.y); A.u[3] = pk2(q1.z, q1.w);
            A.u[4] = pk2(q2.x, q2.y); A.u[5] = pk2(q2.z, q2.w);
            A.u[6] = pk2(q3.x, q3.y); A.u[7] = pk2(q3.z, q3.w);

            const unsigned short* bk = brow + kc * 32;
            #pragma unroll
            for (int nt = 0; nt < 4; ++nt) {
                union { v16bf v; uint4 u[2]; } B;
                B.u[0] = *(const uint4*)(bk + nt * 16 * W1_STR);
                B.u[1] = *(const uint4*)(bk + nt * 16 * W1_STR + 16);
                acc[nt] = __builtin_amdgcn_wmma_f32_16x16x32_bf16(
                            false, A.v, false, B.v, (short)0, acc[nt],
                            false, false);
            }
        }

        // epilogue: bias + relu + W2 partial dot, fused on accumulators
        float partial[8];
        #pragma unroll
        for (int r = 0; r < 8; ++r) partial[r] = 0.f;
        #pragma unroll
        for (int nt = 0; nt < 4; ++nt) {
            const int n = nt * 16 + lsub;
            const float b1n = b1[n], w2n = W2[n];
            #pragma unroll
            for (int r = 0; r < 8; ++r) {
                float hv = fmaxf(acc[nt][r] + b1n, 0.f);
                partial[r] += hv * w2n;
            }
        }
        // reduce N across each 16-lane half-group
        #pragma unroll
        for (int r = 0; r < 8; ++r) {
            float v = partial[r];
            v += __shfl_xor(v, 1, 32);
            v += __shfl_xor(v, 2, 32);
            v += __shfl_xor(v, 4, 32);
            v += __shfl_xor(v, 8, 32);
            partial[r] = v;
        }
        // lanes 0..7 emit rows 0..7, lanes 16..23 emit rows 8..15
        if (lsub < 8) {
            const int m  = lsub + hi * 8;
            const int e2 = eb + m;
            const float z    = partial[lsub] + b2v;
            const float gate = 1.f / (1.f + __expf(-z));
            const int   s2   = edges[3 * e2];
            const float sc   = fmaxf(gate * evidence[s2], 1e-8f);
            keys[e2] = ((u64)__float_as_uint(sc) << 32)
                     | (unsigned)(~(unsigned)e2);
        }
    }
}

// ---------------------------------------------------------------------------
// Top-k: pad keys, bitonic sort descending (exact top_k incl. tie order).
// LDS-fused phases: any compare with j <= SORT_TILE/2 stays inside a
// 4096-element, 4096-aligned chunk -> done in LDS with barriers.
// ---------------------------------------------------------------------------
__global__ void pad_keys_kernel(u64* __restrict__ keys) {
    int i = blockIdx.x * blockDim.x + threadIdx.x;
    if (i < NPAD - N_EDGES) keys[N_EDGES + i] = 0ull;
}

__device__ __forceinline__ void bitonic_phase_lds(u64* s, unsigned base,
                                                  unsigned j, unsigned k) {
    // SORT_TILE/2 comparisons, strided over the block's threads
    for (unsigned t = threadIdx.x; t < SORT_TILE / 2; t += blockDim.x) {
        unsigned i = ((t & ~(j - 1)) << 1) | (t & (j - 1));
        unsigned p = i + j;
        bool up = (((base + i) & k) == 0);      // descending sort
        u64 a = s[i], b = s[p];
        if (up ? (a < b) : (a > b)) { s[i] = b; s[p] = a; }
    }
}

// All stages k = 2..SORT_TILE fused: produces 4096-wide bitonic-sorted runs.
__global__ __launch_bounds__(256) void bitonic_local_sort_kernel(
    u64* __restrict__ keys) {
    __shared__ u64 s[SORT_TILE];
    const unsigned base = blockIdx.x * SORT_TILE;
    for (unsigned t = threadIdx.x; t < SORT_TILE; t += blockDim.x)
        s[t] = keys[base + t];
    __syncthreads();
    for (unsigned kk = 2; kk <= SORT_TILE; kk <<= 1)
        for (unsigned j = kk >> 1; j >= 1; j >>= 1) {
            bitonic_phase_lds(s, base, j, kk);
            __syncthreads();
        }
    for (unsigned t = threadIdx.x; t < SORT_TILE; t += blockDim.x)
        keys[base + t] = s[t];
}

// Tail of stage k (all phases j <= SORT_TILE/2) fused in LDS.
__global__ __launch_bounds__(256) void bitonic_fused_tail_kernel(
    u64* __restrict__ keys, unsigned k) {
    __shared__ u64 s[SORT_TILE];
    const unsigned base = blockIdx.x * SORT_TILE;
    for (unsigned t = threadIdx.x; t < SORT_TILE; t += blockDim.x)
        s[t] = keys[base + t];
    __syncthreads();
    for (unsigned j = SORT_TILE / 2; j >= 1; j >>= 1) {
        bitonic_phase_lds(s, base, j, k);
        __syncthreads();
    }
    for (unsigned t = threadIdx.x; t < SORT_TILE; t += blockDim.x)
        keys[base + t] = s[t];
}

// Global phase (j >= SORT_TILE): partner is in another chunk.
__global__ void bitonic_step_kernel(u64* __restrict__ keys,
                                    unsigned j, unsigned k) {
    unsigned i = blockIdx.x * blockDim.x + threadIdx.x;
    unsigned ixj = i ^ j;
    if (ixj > i && i < NPAD) {
        u64 va = keys[i], vb = keys[ixj];
        bool up = ((i & k) == 0);               // descending sort
        if (up ? (va < vb) : (va > vb)) { keys[i] = vb; keys[ixj] = va; }
    }
}

// ---------------------------------------------------------------------------
// Outputs
// ---------------------------------------------------------------------------
__global__ void zero_ev_kernel(float* __restrict__ out_ev,
                               unsigned* __restrict__ maxbits) {
    int i = blockIdx.x * blockDim.x + threadIdx.x;
    if (i < N_NODES) out_ev[i] = 0.f;
    if (i == 0) *maxbits = 0u;
}

__global__ void emit_scatter_kernel(const u64* __restrict__ keys,
                                    const int* __restrict__ edges,
                                    const int* __restrict__ ebi,
                                    const float* __restrict__ evidence,
                                    int* __restrict__ out_edges,
                                    int* __restrict__ out_batch,
                                    float* __restrict__ out_w,
                                    float* __restrict__ out_ev) {
    int i = blockIdx.x * blockDim.x + threadIdx.x;
    if (i >= KEEP) return;
    u64 key = keys[i];
    unsigned e = ~(unsigned)key;
    float    w = __uint_as_float((unsigned)(key >> 32));
    int sub = edges[3 * e + 0], rel = edges[3 * e + 1], obj = edges[3 * e + 2];
    out_edges[3 * i + 0] = sub;
    out_edges[3 * i + 1] = rel;
    out_edges[3 * i + 2] = obj;
    out_batch[i] = ebi[e];
    out_w[i]     = w;
    atomicAdd(&out_ev[obj], evidence[sub] * w);   // segment_sum
}

__global__ void max_ev_kernel(const float* __restrict__ out_ev,
                              unsigned* __restrict__ maxbits) {
    int i = blockIdx.x * blockDim.x + threadIdx.x;
    if (i < N_NODES) {
        float v = fmaxf(out_ev[i], 0.f);          // sums are >= 0
        atomicMax(maxbits, __float_as_uint(v));   // bit-monotone for >= 0
    }
}

__global__ void norm_ev_kernel(float* __restrict__ out_ev,
                               const unsigned* __restrict__ maxbits) {
    int i = blockIdx.x * blockDim.x + threadIdx.x;
    if (i >= N_NODES) return;
    float maxv = __uint_as_float(*maxbits);
    if (maxv > 1e-12f) out_ev[i] = out_ev[i] / (maxv + 1e-12f);
}

// ---------------------------------------------------------------------------
// Launch
// ---------------------------------------------------------------------------
extern "C" void kernel_launch(void* const* d_in, const int* in_sizes, int n_in,
                              void* d_out, int out_size, void* d_ws, size_t ws_size,
                              hipStream_t stream) {
    const float* hidden    = (const float*)d_in[0];
    const int*   edges     = (const int*)  d_in[1];
    const int*   q_rel     = (const int*)  d_in[2];
    const int*   ebi       = (const int*)  d_in[3];
    const float* evidence  = (const float*)d_in[4];
    const float* rel_embed = (const float*)d_in[5];
    const float* W1        = (const float*)d_in[6];
    const float* b1        = (const float*)d_in[7];
    const float* W2        = (const float*)d_in[8];
    const float* b2        = (const float*)d_in[9];

    u64* keys = (u64*)d_ws;                                    // 8 MB
    unsigned* maxbits = (unsigned*)((char*)d_ws + (size_t)NPAD * 8);

    int*   out_edges = (int*)d_out;
    int*   out_batch = (int*)d_out + OFF_BATCH;
    float* out_w     = (float*)d_out + OFF_W;
    float* out_ev    = (float*)d_out + OFF_EV;

    // 1) MLP gate scores -> sort keys (WMMA path)
    score_wmma_kernel<<<512, 256, 0, stream>>>(
        hidden, edges, q_rel, ebi, evidence, rel_embed, W1, b1, W2, b2, keys);

    // 2) pad to power of two
    pad_keys_kernel<<<(NPAD - N_EDGES + 255) / 256, 256, 0, stream>>>(keys);

    // 3) bitonic sort, descending; LDS-fused phases (45 launches total)
    bitonic_local_sort_kernel<<<NPAD / SORT_TILE, 256, 0, stream>>>(keys);
    for (unsigned k = 2u * SORT_TILE; k <= (unsigned)NPAD; k <<= 1) {
        for (unsigned j = k >> 1; j >= (unsigned)SORT_TILE; j >>= 1)
            bitonic_step_kernel<<<NPAD / 256, 256, 0, stream>>>(keys, j, k);
        bitonic_fused_tail_kernel<<<NPAD / SORT_TILE, 256, 0, stream>>>(keys, k);
    }

    // 4) outputs
    zero_ev_kernel<<<(N_NODES + 255) / 256, 256, 0, stream>>>(out_ev, maxbits);
    emit_scatter_kernel<<<(KEEP + 255) / 256, 256, 0, stream>>>(
        keys, edges, ebi, evidence, out_edges, out_batch, out_w, out_ev);
    max_ev_kernel<<<(N_NODES + 255) / 256, 256, 0, stream>>>(out_ev, maxbits);
    norm_ev_kernel<<<(N_NODES + 255) / 256, 256, 0, stream>>>(out_ev, maxbits);
}
